// Attention_72224170050112
// MI455X (gfx1250) — matrix operand order repaired and checked by reading, not compiled, not run on hardware
//
#include <hip/hip_runtime.h>
#include <hip/hip_bf16.h>

// ---------------- problem constants ----------------
#define BB 4
#define TT 2048
#define DD 1024
#define HH 16
#define HD 64
#define N3 3072
#define MM (BB*TT)   // 8192

// ---------------- types ----------------
typedef __attribute__((ext_vector_type(16))) __bf16        bf16x16;
typedef __attribute__((ext_vector_type(8)))  float         f32x8;
typedef __attribute__((ext_vector_type(4)))  unsigned int  u32x4;
typedef __attribute__((ext_vector_type(2)))  unsigned int  u32x2;
typedef __attribute__((ext_vector_type(2)))  int           i32x2;

// address-space-qualified pointee types for the async-LDS builtin
typedef __attribute__((address_space(1))) i32x2 gi32x2;   // global
typedef __attribute__((address_space(3))) i32x2 li32x2;   // LDS

union Frag {
  bf16x16 v;
  u32x4   q[2];
};

__device__ __forceinline__ f32x8 wmma_bf16(const Frag& a, const Frag& b, f32x8 c) {
  // (neg_a, A, neg_b, B, c_mod, C, reuse_a, reuse_b)
  return __builtin_amdgcn_wmma_f32_16x16x32_bf16(false, a.v, false, b.v, (short)0, c, false, false);
}

// -------- async global -> LDS (CDNA5 GLOBAL_LOAD_ASYNC_TO_LDS, ASYNCcnt) --------
#if defined(__has_builtin)
#if __has_builtin(__builtin_amdgcn_global_load_async_to_lds_b64) && \
    __has_builtin(__builtin_amdgcn_s_wait_asynccnt)
#define HAVE_ASYNC_LDS 1
#endif
#endif
#ifndef HAVE_ASYNC_LDS
#define HAVE_ASYNC_LDS 0
#endif

__device__ __forceinline__ void async_copy_b64(const void* g, void* l) {
#if HAVE_ASYNC_LDS
  __builtin_amdgcn_global_load_async_to_lds_b64((gi32x2*)g, (li32x2*)l, 0, 0);
#else
  *(u32x2*)l = *(const u32x2*)g;
#endif
}
__device__ __forceinline__ void async_wait0() {
#if HAVE_ASYNC_LDS
  __builtin_amdgcn_s_wait_asynccnt(0);
#endif
}

// ---------------- fp32 -> bf16 convert ----------------
__global__ void f32_to_bf16(const float* __restrict__ in, __bf16* __restrict__ out, int n) {
  int i = blockIdx.x * blockDim.x + threadIdx.x;
  int stride = gridDim.x * blockDim.x;
  for (; i < n; i += stride) out[i] = (__bf16)in[i];
}

// ---------------- QKV GEMM: X[M,K] @ W[K,3D] + b, scatter to Q/K/V [B,H,T,HD] ----------------
__global__ __launch_bounds__(256)
void qkv_gemm(const __bf16* __restrict__ X, const __bf16* __restrict__ Wq,
              const float* __restrict__ bias,
              __bf16* __restrict__ Qo, __bf16* __restrict__ Ko, __bf16* __restrict__ Vo) {
  __shared__ __align__(16) __bf16 As[128][72];   // [m][k], 144B row stride (16B multiple)
  __shared__ __align__(16) __bf16 Bt[128][72];   // [n][k], K-contiguous per column

  const int tid  = threadIdx.x;
  const int lane = tid & 31, wid = tid >> 5;
  const int wm = wid & 3, wn = wid >> 2;          // 4 x 2 wave grid
  const int g = lane >> 4, ln = lane & 15;
  const int m0 = blockIdx.y * 128;
  const int n0 = blockIdx.x * 128;

  f32x8 acc[2][4] = {};

  for (int k0 = 0; k0 < DD; k0 += 64) {
    // stage A tile (128x64) via async global->LDS (no VGPR round-trip)
    #pragma unroll
    for (int i = 0; i < 8; ++i) {
      int c = tid + i * 256;
      int row = c >> 4, kc = (c & 15) * 4;
      async_copy_b64(&X[(size_t)(m0 + row) * DD + k0 + kc], &As[row][kc]);
    }
    // stage B tile (64x128), transposed into Bt[n][k]
    #pragma unroll
    for (int i = 0; i < 8; ++i) {
      int c = tid + i * 256;
      int kr = c >> 5, nc = (c & 31) * 4;
      u32x2 d = *(const u32x2*)&Wq[(size_t)(k0 + kr) * N3 + n0 + nc];
      __bf16 t4[4]; *(u32x2*)t4 = d;
      Bt[nc + 0][kr] = t4[0]; Bt[nc + 1][kr] = t4[1];
      Bt[nc + 2][kr] = t4[2]; Bt[nc + 3][kr] = t4[3];
    }
    if (k0 + 64 < DD) {
      __builtin_prefetch(&X[(size_t)(m0 + (tid >> 4)) * DD + k0 + 64], 0, 1);
      __builtin_prefetch(&Wq[(size_t)(k0 + 64 + (tid >> 5)) * N3 + n0 + (tid & 31) * 4], 0, 1);
    }
    async_wait0();
    __syncthreads();

    #pragma unroll
    for (int kk = 0; kk < 64; kk += 32) {
      Frag a[2], b[4];
      #pragma unroll
      for (int mi = 0; mi < 2; ++mi) {
        int row = wm * 32 + mi * 16 + ln;
        a[mi].q[0] = *(const u32x4*)&As[row][kk + 8 * g];
        a[mi].q[1] = *(const u32x4*)&As[row][kk + 16 + 8 * g];
      }
      #pragma unroll
      for (int ni = 0; ni < 4; ++ni) {
        int col = wn * 64 + ni * 16 + ln;
        b[ni].q[0] = *(const u32x4*)&Bt[col][kk + 8 * g];
        b[ni].q[1] = *(const u32x4*)&Bt[col][kk + 16 + 8 * g];
      }
      #pragma unroll
      for (int mi = 0; mi < 2; ++mi)
        #pragma unroll
        for (int ni = 0; ni < 4; ++ni)
          acc[mi][ni] = wmma_bf16(a[mi], b[ni], acc[mi][ni]);
    }
    __syncthreads();
  }

  // epilogue: bias + scatter into per-head layout
  #pragma unroll
  for (int mi = 0; mi < 2; ++mi) {
    #pragma unroll
    for (int ni = 0; ni < 4; ++ni) {
      int col = n0 + wn * 64 + ni * 16 + ln;
      float bv = bias[col];
      int which = col >> 10;
      int rem = col & 1023;
      int h = rem >> 6, dd = rem & 63;
      __bf16* base = (which == 0) ? Qo : (which == 1) ? Ko : Vo;
      #pragma unroll
      for (int r = 0; r < 8; ++r) {
        int m = m0 + wm * 32 + mi * 16 + r + 8 * g;
        int bidx = m >> 11, t = m & 2047;
        size_t idx = ((size_t)((bidx * HH + h) * TT + t)) * HD + dd;
        base[idx] = (__bf16)(acc[mi][ni][r] + bv);
      }
    }
  }
}

// ---------------- flash attention per (b,h), q-tile=64, key-block=64 ----------------
__global__ __launch_bounds__(128)
void flash_attn(const __bf16* __restrict__ Qg, const __bf16* __restrict__ Kg,
                const __bf16* __restrict__ Vg, __bf16* __restrict__ Og) {
  __shared__ __align__(16) __bf16 Qs[64][72];       // [q][hd]
  __shared__ __align__(16) __bf16 Ks[64][72];       // [key][hd]  (B-frag friendly for QK^T)
  __shared__ __align__(16) __bf16 Vt[64][72];       // [hd][key]  (B-frag friendly for PV)
  __shared__ __align__(16) __bf16 Pl[4][16][72];    // per-wave P staging (C -> A layout)

  const int tid  = threadIdx.x;
  const int lane = tid & 31, w = tid >> 5;
  const int g = lane >> 4, ln = lane & 15;
  const int q0 = blockIdx.x * 64;
  const int bh = blockIdx.y;
  const size_t hb = (size_t)bh * TT * HD;

  // load Q tile (async; completion covered by first iteration's async_wait+barrier)
  #pragma unroll
  for (int i = 0; i < 8; ++i) {
    int c = tid + i * 128;
    int row = c >> 4, kc = (c & 15) * 4;
    async_copy_b64(&Qg[hb + (size_t)(q0 + row) * HD + kc], &Qs[row][kc]);
  }

  float mrow[8], lrow[8];
  #pragma unroll
  for (int r = 0; r < 8; ++r) { mrow[r] = -1e30f; lrow[r] = 0.f; }
  f32x8 o[4] = {};

  for (int kb = 0; kb < TT / 64; ++kb) {
    __syncthreads();     // previous block's Ks/Vt consumers done
    const int kt0 = kb * 64;
    #pragma unroll
    for (int i = 0; i < 8; ++i) {
      int c = tid + i * 128;
      int row = c >> 4, kc = (c & 15) * 4;
      async_copy_b64(&Kg[hb + (size_t)(kt0 + row) * HD + kc], &Ks[row][kc]);
      u32x2 d = *(const u32x2*)&Vg[hb + (size_t)(kt0 + row) * HD + kc];
      __bf16 t4[4]; *(u32x2*)t4 = d;
      Vt[kc + 0][row] = t4[0]; Vt[kc + 1][row] = t4[1];
      Vt[kc + 2][row] = t4[2]; Vt[kc + 3][row] = t4[3];
    }
    async_wait0();
    __syncthreads();

    // S = Q K^T  (contraction over HD=64, two K=32 steps)
    const int qrow = w * 16 + ln;
    Frag a0, a1;
    a0.q[0] = *(const u32x4*)&Qs[qrow][8 * g];
    a0.q[1] = *(const u32x4*)&Qs[qrow][16 + 8 * g];
    a1.q[0] = *(const u32x4*)&Qs[qrow][32 + 8 * g];
    a1.q[1] = *(const u32x4*)&Qs[qrow][48 + 8 * g];

    f32x8 s[4];
    #pragma unroll
    for (int ni = 0; ni < 4; ++ni) {
      int key = ni * 16 + ln;
      Frag b0, b1;
      b0.q[0] = *(const u32x4*)&Ks[key][8 * g];
      b0.q[1] = *(const u32x4*)&Ks[key][16 + 8 * g];
      b1.q[0] = *(const u32x4*)&Ks[key][32 + 8 * g];
      b1.q[1] = *(const u32x4*)&Ks[key][48 + 8 * g];
      f32x8 z = {};
      z = wmma_bf16(a0, b0, z);
      s[ni] = wmma_bf16(a1, b1, z);
    }

    // online softmax (rows live in 16-lane groups; xor masks <=8 stay in-group)
    const float sc = 0.125f;   // 1/sqrt(64)
    #pragma unroll
    for (int r = 0; r < 8; ++r) {
      float v0 = s[0][r] * sc, v1 = s[1][r] * sc, v2 = s[2][r] * sc, v3 = s[3][r] * sc;
      float mx = fmaxf(fmaxf(v0, v1), fmaxf(v2, v3));
      mx = fmaxf(mx, __shfl_xor(mx, 1, 32));
      mx = fmaxf(mx, __shfl_xor(mx, 2, 32));
      mx = fmaxf(mx, __shfl_xor(mx, 4, 32));
      mx = fmaxf(mx, __shfl_xor(mx, 8, 32));
      float mnew  = fmaxf(mrow[r], mx);
      float alpha = __expf(mrow[r] - mnew);
      mrow[r] = mnew;
      float p0 = __expf(v0 - mnew), p1 = __expf(v1 - mnew);
      float p2 = __expf(v2 - mnew), p3 = __expf(v3 - mnew);
      float rs = p0 + p1 + p2 + p3;
      rs += __shfl_xor(rs, 1, 32);
      rs += __shfl_xor(rs, 2, 32);
      rs += __shfl_xor(rs, 4, 32);
      rs += __shfl_xor(rs, 8, 32);
      lrow[r] = lrow[r] * alpha + rs;
      o[0][r] *= alpha; o[1][r] *= alpha; o[2][r] *= alpha; o[3][r] *= alpha;
      int pr = r + 8 * g;
      Pl[w][pr][ 0 + ln] = (__bf16)p0;
      Pl[w][pr][16 + ln] = (__bf16)p1;
      Pl[w][pr][32 + ln] = (__bf16)p2;
      Pl[w][pr][48 + ln] = (__bf16)p3;
    }

    // O += P V  (contraction over 64 keys, two K=32 steps)
    Frag pa0, pa1;
    pa0.q[0] = *(const u32x4*)&Pl[w][ln][8 * g];
    pa0.q[1] = *(const u32x4*)&Pl[w][ln][16 + 8 * g];
    pa1.q[0] = *(const u32x4*)&Pl[w][ln][32 + 8 * g];
    pa1.q[1] = *(const u32x4*)&Pl[w][ln][48 + 8 * g];
    #pragma unroll
    for (int nj = 0; nj < 4; ++nj) {
      int hd = nj * 16 + ln;
      Frag vb0, vb1;
      vb0.q[0] = *(const u32x4*)&Vt[hd][8 * g];
      vb0.q[1] = *(const u32x4*)&Vt[hd][16 + 8 * g];
      vb1.q[0] = *(const u32x4*)&Vt[hd][32 + 8 * g];
      vb1.q[1] = *(const u32x4*)&Vt[hd][48 + 8 * g];
      o[nj] = wmma_bf16(pa0, vb0, o[nj]);
      o[nj] = wmma_bf16(pa1, vb1, o[nj]);
    }
  }

  // normalize + write back in [B,T,D] bf16 layout (fuses head transpose)
  const int bidx = bh >> 4;
  const int h    = bh & 15;
  #pragma unroll
  for (int nj = 0; nj < 4; ++nj) {
    #pragma unroll
    for (int r = 0; r < 8; ++r) {
      int t = q0 + w * 16 + r + 8 * g;
      float val = o[nj][r] / lrow[r];
      size_t idx = ((size_t)(bidx * TT + t)) * DD + h * HD + nj * 16 + ln;
      Og[idx] = (__bf16)val;
    }
  }
}

// ---------------- projection GEMM: A[M,D] @ W[D,D] + b -> f32 out ----------------
__global__ __launch_bounds__(256)
void proj_gemm(const __bf16* __restrict__ A, const __bf16* __restrict__ W,
               const float* __restrict__ bias, float* __restrict__ out) {
  __shared__ __align__(16) __bf16 As[128][72];
  __shared__ __align__(16) __bf16 Bt[128][72];

  const int tid  = threadIdx.x;
  const int lane = tid & 31, wid = tid >> 5;
  const int wm = wid & 3, wn = wid >> 2;
  const int g = lane >> 4, ln = lane & 15;
  const int m0 = blockIdx.y * 128;
  const int n0 = blockIdx.x * 128;

  f32x8 acc[2][4] = {};

  for (int k0 = 0; k0 < DD; k0 += 64) {
    #pragma unroll
    for (int i = 0; i < 8; ++i) {
      int c = tid + i * 256;
      int row = c >> 4, kc = (c & 15) * 4;
      async_copy_b64(&A[(size_t)(m0 + row) * DD + k0 + kc], &As[row][kc]);
    }
    #pragma unroll
    for (int i = 0; i < 8; ++i) {
      int c = tid + i * 256;
      int kr = c >> 5, nc = (c & 31) * 4;
      u32x2 d = *(const u32x2*)&W[(size_t)(k0 + kr) * DD + n0 + nc];
      __bf16 t4[4]; *(u32x2*)t4 = d;
      Bt[nc + 0][kr] = t4[0]; Bt[nc + 1][kr] = t4[1];
      Bt[nc + 2][kr] = t4[2]; Bt[nc + 3][kr] = t4[3];
    }
    if (k0 + 64 < DD) {
      __builtin_prefetch(&A[(size_t)(m0 + (tid >> 4)) * DD + k0 + 64], 0, 1);
      __builtin_prefetch(&W[(size_t)(k0 + 64 + (tid >> 5)) * DD + n0 + (tid & 31) * 4], 0, 1);
    }
    async_wait0();
    __syncthreads();

    #pragma unroll
    for (int kk = 0; kk < 64; kk += 32) {
      Frag a[2], b[4];
      #pragma unroll
      for (int mi = 0; mi < 2; ++mi) {
        int row = wm * 32 + mi * 16 + ln;
        a[mi].q[0] = *(const u32x4*)&As[row][kk + 8 * g];
        a[mi].q[1] = *(const u32x4*)&As[row][kk + 16 + 8 * g];
      }
      #pragma unroll
      for (int ni = 0; ni < 4; ++ni) {
        int col = wn * 64 + ni * 16 + ln;
        b[ni].q[0] = *(const u32x4*)&Bt[col][kk + 8 * g];
        b[ni].q[1] = *(const u32x4*)&Bt[col][kk + 16 + 8 * g];
      }
      #pragma unroll
      for (int mi = 0; mi < 2; ++mi)
        #pragma unroll
        for (int ni = 0; ni < 4; ++ni)
          acc[mi][ni] = wmma_bf16(a[mi], b[ni], acc[mi][ni]);
    }
    __syncthreads();
  }

  #pragma unroll
  for (int mi = 0; mi < 2; ++mi) {
    #pragma unroll
    for (int ni = 0; ni < 4; ++ni) {
      int col = n0 + wn * 64 + ni * 16 + ln;
      float bv = bias[col];
      #pragma unroll
      for (int r = 0; r < 8; ++r) {
        int m = m0 + wm * 32 + mi * 16 + r + 8 * g;
        out[(size_t)m * DD + col] = acc[mi][ni][r] + bv;
      }
    }
  }
}

// ---------------- host launcher ----------------
extern "C" void kernel_launch(void* const* d_in, const int* in_sizes, int n_in,
                              void* d_out, int out_size, void* d_ws, size_t ws_size,
                              hipStream_t stream) {
  const float* x      = (const float*)d_in[0];
  const float* w_qkv  = (const float*)d_in[1];
  const float* b_qkv  = (const float*)d_in[2];
  const float* w_proj = (const float*)d_in[3];
  const float* b_proj = (const float*)d_in[4];
  float* out = (float*)d_out;

  char* ws = (char*)d_ws;
  size_t off = 0;
  auto wsalloc = [&](size_t bytes) -> char* {
    char* p = ws + off;
    off = (off + bytes + 255) & ~(size_t)255;
    return p;
  };

  __bf16* xbf  = (__bf16*)wsalloc((size_t)MM * DD * 2);   // also reused for attn output
  __bf16* wqbf = (__bf16*)wsalloc((size_t)DD * N3 * 2);
  __bf16* wpbf = (__bf16*)wsalloc((size_t)DD * DD * 2);
  __bf16* Qb   = (__bf16*)wsalloc((size_t)MM * DD * 2);
  __bf16* Kb   = (__bf16*)wsalloc((size_t)MM * DD * 2);
  __bf16* Vb   = (__bf16*)wsalloc((size_t)MM * DD * 2);
  __bf16* attn = xbf;  // x no longer needed after qkv_gemm; alias to save workspace

  f32_to_bf16<<<2048, 256, 0, stream>>>(x,      xbf,  MM * DD);
  f32_to_bf16<<<1024, 256, 0, stream>>>(w_qkv,  wqbf, DD * N3);
  f32_to_bf16<<<512,  256, 0, stream>>>(w_proj, wpbf, DD * DD);

  qkv_gemm<<<dim3(N3 / 128, MM / 128), 256, 0, stream>>>(xbf, wqbf, b_qkv, Qb, Kb, Vb);
  flash_attn<<<dim3(TT / 64, BB * HH), 128, 0, stream>>>(Qb, Kb, Vb, attn);
  proj_gemm<<<dim3(DD / 128, MM / 128), 256, 0, stream>>>(attn, wpbf, b_proj, out);
}
